// GCNPool_my2_44727789421337
// MI455X (gfx1250) — matrix operand order repaired and checked by reading, not compiled, not run on hardware
//
#include <hip/hip_runtime.h>
#include <hip/hip_bf16.h>

typedef __attribute__((ext_vector_type(16))) _Float16 v16h;
typedef __attribute__((ext_vector_type(8)))  float    v8f;
typedef __attribute__((ext_vector_type(4)))  unsigned int u32x4;
typedef __attribute__((ext_vector_type(8)))  unsigned int u32x8;

#define DEV __device__ __forceinline__

DEV v8f wmma_f16(v16h a, v16h b, v8f c) {
  return __builtin_amdgcn_wmma_f32_16x16x32_f16(false, a, false, b, (short)0, c, false, false);
}
// A-matrix K index for element i of the v16h operand (contiguous k = 16*hf + i, matching B)
DEV int a_k(int hf, int i) { return 16 * hf + i; }

// ---------------- constants ----------------
#define BB 16
#define NN 500
#define TT_ 10
#define NROW 80000      // B*N*T
#define NP 512          // padded N for GCN branch
#define NSLICE 160      // B*T

// ---------------- workspace offsets (bytes) ----------------
static constexpr size_t OFF_RES = 0;                       // (b,n,t,32) f32  10,240,000
static constexpr size_t OFF_W1  = OFF_RES + 10240000;      // h1 -> o_att -> g2   20,480,000
static constexpr size_t OFF_W2  = OFF_W1 + 20480000;       // q -> h2             20,480,000
static constexpr size_t OFF_W3  = OFF_W2 + 20480000;       // k -> h3             20,480,000
static constexpr size_t OFF_W4  = OFF_W3 + 20480000;       // v -> g -> xo_pre    20,480,000
static constexpr size_t OFF_XTH = OFF_W4 + 20480000;       // f16 (bt,512,32)  5,242,880
static constexpr size_t OFF_AH  = OFF_XTH + 5242880;       // f16 512x512        524,288
static constexpr size_t OFF_Z1  = OFF_AH + 524288;
static constexpr size_t OFF_Z2  = OFF_Z1 + 5242880;
static constexpr size_t OFF_Z3  = OFF_Z2 + 5242880;
static constexpr size_t OFF_Z4  = OFF_Z3 + 5242880;
static constexpr size_t OFF_F1  = OFF_Z4 + 5242880;        // (B,6,500) f32 192,000
static constexpr size_t OFF_F2  = OFF_F1 + 192000;         // (B,32,6)  12,288
static constexpr size_t OFF_M1  = OFF_F2 + 12288;          // (B,6,32)  12,288
static constexpr size_t OFF_TC  = OFF_M1 + 12288;          // (B,6,6)   2,304
static constexpr size_t OFF_ACC = OFF_TC + 2304 + 64;      // 64 floats

// ======== 1x1 conv from x (B,C=32,N,T) layout: out[(b,n,t),o] = sum_c x*W[o,c]+b ========
__global__ void k_c1x1_x(const float* __restrict__ x, const float* __restrict__ W,
                         const float* __restrict__ bias, float* __restrict__ out, int Cout) {
  const int lane = threadIdx.x, hf = lane >> 4, lm = lane & 15;
  const int rowbase = blockIdx.x * 16;
  const int row = rowbase + lm;
  const int b = row / 5000, rem = row % 5000, n = rem / 10, t = rem % 10;
  v16h a;
#pragma unroll
  for (int i = 0; i < 16; ++i) {
    int c = a_k(hf, i);
    a[i] = (_Float16)x[((size_t)(b * 32 + c) * 500 + n) * 10 + t];
  }
  for (int ob = 0; ob < Cout; ob += 16) {
    v16h bm;
#pragma unroll
    for (int i = 0; i < 16; ++i) bm[i] = (_Float16)W[(ob + lm) * 32 + 16 * hf + i];
    v8f c = {};
    c = wmma_f16(a, bm, c);
    float bv = bias[ob + lm];
#pragma unroll
    for (int r = 0; r < 8; ++r)
      out[(size_t)(rowbase + r + 8 * hf) * Cout + ob + lm] = c[r] + bv;
  }
}

// ======== temporal conv k=3 (pad 1): K=192 = 3 taps x 64 ch, Cout=64 ========
__global__ void k_convt3(const float* __restrict__ h, const float* __restrict__ W,
                         const float* __restrict__ bias, float* __restrict__ out) {
  const int lane = threadIdx.x, hf = lane >> 4, lm = lane & 15;
  const int rowbase = blockIdx.x * 16;
  const int row = rowbase + lm;
  const int bn = row / 10, t = row % 10;
  v8f acc[4] = {};
  for (int ks = 0; ks < 6; ++ks) {
    v16h a;
#pragma unroll
    for (int i = 0; i < 16; ++i) {
      int kk = 32 * ks + a_k(hf, i);
      int tap = kk >> 6, c = kk & 63, tt = t + tap - 1;
      float v = (tt >= 0 && tt < 10) ? h[((size_t)bn * 10 + tt) * 64 + c] : 0.0f;
      a[i] = (_Float16)v;
    }
    for (int nt = 0; nt < 4; ++nt) {
      v16h bm;
#pragma unroll
      for (int i = 0; i < 16; ++i) {
        int kk = 32 * ks + 16 * hf + i;
        int tap = kk >> 6, c = kk & 63;
        bm[i] = (_Float16)W[((nt * 16 + lm) * 64 + c) * 3 + tap];
      }
      acc[nt] = wmma_f16(a, bm, acc[nt]);
    }
  }
  for (int nt = 0; nt < 4; ++nt) {
    float bv = bias[nt * 16 + lm];
#pragma unroll
    for (int r = 0; r < 8; ++r)
      out[(size_t)(rowbase + r + 8 * hf) * 64 + nt * 16 + lm] = acc[nt][r] + bv;
  }
}

// ======== plain row-major GEMM: out[(row),o] = sum_k in[row,k]*W[o,k]+b[o] ========
__global__ void k_gemm_rm(const float* __restrict__ in, const float* __restrict__ W,
                          const float* __restrict__ bias, float* __restrict__ out,
                          int K, int Cout) {
  const int lane = threadIdx.x, hf = lane >> 4, lm = lane & 15;
  const size_t rowbase = (size_t)blockIdx.x * 16;
  for (int ob = 0; ob < Cout; ob += 16) {
    v8f c = {};
    for (int ks = 0; ks < K; ks += 32) {
      v16h a, bm;
#pragma unroll
      for (int i = 0; i < 16; ++i) {
        a[i]  = (_Float16)in[(rowbase + lm) * K + ks + a_k(hf, i)];
        bm[i] = (_Float16)W[(ob + lm) * K + ks + 16 * hf + i];
      }
      c = wmma_f16(a, bm, c);
    }
    float bv = bias[ob + lm];
#pragma unroll
    for (int r = 0; r < 8; ++r)
      out[(rowbase + r + 8 * hf) * Cout + ob + lm] = c[r] + bv;
  }
}

// ======== scalar multi-head self-attention, T=10, dk=8, NH=8 ========
__global__ void k_mhsa(const float* __restrict__ q, const float* __restrict__ k,
                       const float* __restrict__ v, float* __restrict__ o) {
  int idx = blockIdx.x * blockDim.x + threadIdx.x;
  if (idx >= BB * NN * 8 * 10) return;
  int t = idx % 10, h = (idx / 10) % 8, bn = idx / 80;
  const float* qp = q + ((size_t)bn * 10 + t) * 64 + h * 8;
  float qv[8];
#pragma unroll
  for (int d = 0; d < 8; ++d) qv[d] = qp[d];
  float sc[10], mx = -1e30f;
  for (int s = 0; s < 10; ++s) {
    const float* kp = k + ((size_t)bn * 10 + s) * 64 + h * 8;
    float a = 0;
#pragma unroll
    for (int d = 0; d < 8; ++d) a += qv[d] * kp[d];
    sc[s] = a * 0.35355339059327373f;
    mx = fmaxf(mx, sc[s]);
  }
  float sum = 0;
  for (int s = 0; s < 10; ++s) { sc[s] = __expf(sc[s] - mx); sum += sc[s]; }
  float inv = 1.0f / sum;
  float ov[8] = {};
  for (int s = 0; s < 10; ++s) {
    const float* vp = v + ((size_t)bn * 10 + s) * 64 + h * 8;
    float p = sc[s] * inv;
#pragma unroll
    for (int d = 0; d < 8; ++d) ov[d] += p * vp[d];
  }
  float* op = o + ((size_t)bn * 10 + t) * 64 + h * 8;
#pragma unroll
  for (int d = 0; d < 8; ++d) op[d] = ov[d];
}

// ======== build xt f16 (bt,512,32), zero pad rows ========
__global__ void k_make_xth(const float* __restrict__ h3, _Float16* __restrict__ xth) {
  int idx = blockIdx.x * blockDim.x + threadIdx.x;
  if (idx >= NSLICE * NP * 32) return;
  int c = idx & 31, n = (idx >> 5) & 511, bt = idx >> 14;
  int b = bt / 10, t = bt % 10;
  float v = (n < 500) ? h3[(((size_t)b * 500 + n) * 10 + t) * 32 + c] : 0.0f;
  xth[idx] = (_Float16)v;
}

__global__ void k_make_ah(const float* __restrict__ A, _Float16* __restrict__ Ah) {
  int idx = blockIdx.x * blockDim.x + threadIdx.x;
  if (idx >= 512 * 512) return;
  int v = idx & 511, w = idx >> 9;
  Ah[idx] = (w < 500 && v < 500) ? (_Float16)A[(size_t)w * 500 + v] : (_Float16)0.0f;
}

// ======== z[w,c] = sum_v A[w,v]*src[v,c] per slice ========
// One workgroup (8 waves) per slice; the 32 KB src slice is staged into LDS once
// via the Tensor Data Mover (tensor_load_to_lds, TENSORcnt), then all waves run
// WMMA against the LDS copy.
__global__ void k_diffuse(const _Float16* __restrict__ Ah, const _Float16* __restrict__ src,
                          _Float16* __restrict__ dst) {
  __shared__ _Float16 s_src[NP * 32];  // 32 KB slice
  const int tid = threadIdx.x;
  const int lane = tid & 31, hf = lane >> 4, lm = lane & 15;
  const int wave = tid >> 5;  // 0..7
  const size_t sl = (size_t)blockIdx.x * NP * 32;

  if (wave == 0) {
    // Build Tensor DMA descriptor (D#): linear 16384-element (2B) tile -> LDS.
    unsigned long long ga = (unsigned long long)(uintptr_t)(src + sl);
    unsigned int la = (unsigned int)(uintptr_t)(&s_src[0]);  // flat low 32 = LDS addr
    u32x4 g0;
    g0[0] = 1u;                                              // count=1, user mode
    g0[1] = la;                                              // lds_addr
    g0[2] = (unsigned int)ga;                                // global_addr[31:0]
    g0[3] = ((unsigned int)(ga >> 32) & 0x01FFFFFFu) | 0x80000000u;  // addr[56:32] | type=2
    u32x8 g1;
    g1[0] = (1u << 16);          // workgroup_mask=0, data_size=1 (2 bytes)
    g1[1] = (16384u & 0xFFFFu) << 16;   // tensor_dim0[15:0]
    g1[2] = (16384u >> 16) | (1u << 16);// tensor_dim0[31:16] | tensor_dim1[15:0]=1
    g1[3] = (16384u & 0xFFFFu) << 16;   // tensor_dim1[31:16]=0 | tile_dim0=16384
    g1[4] = 1u;                  // tile_dim1=1, tile_dim2=0
    g1[5] = 16384u;              // tensor_dim0_stride[31:0]
    g1[6] = 0u;                  // stride0[47:32] | stride1[15:0]
    g1[7] = 0u;                  // stride1[47:16]
    asm volatile("tensor_load_to_lds %0, %1" :: "s"(g0), "s"(g1) : "memory");
    __builtin_amdgcn_s_wait_tensorcnt(0);
  }
  __syncthreads();

  for (int wt = 0; wt < 4; ++wt) {
    const int wbase = (wave * 4 + wt) * 16;
    v8f a0 = {}, a1 = {};
    for (int ks = 0; ks < NP; ks += 32) {
      v16h a, b0, b1;
#pragma unroll
      for (int i = 0; i < 16; ++i) a[i] = Ah[(size_t)(wbase + lm) * NP + ks + a_k(hf, i)];
#pragma unroll
      for (int i = 0; i < 16; ++i) {
        int k = ks + 16 * hf + i;
        b0[i] = s_src[k * 32 + lm];
        b1[i] = s_src[k * 32 + 16 + lm];
      }
      a0 = wmma_f16(a, b0, a0);
      a1 = wmma_f16(a, b1, a1);
    }
#pragma unroll
    for (int r = 0; r < 8; ++r) {
      int w = wbase + r + 8 * hf;
      dst[sl + (size_t)w * 32 + lm]      = (_Float16)a0[r];
      dst[sl + (size_t)w * 32 + 16 + lm] = (_Float16)a1[r];
    }
  }
}

// ======== GCN attention: scores in LDS, softmax over v<500, then p @ vsrc ========
__global__ void k_gcn_att(const _Float16* __restrict__ xth, const _Float16* __restrict__ vsrc,
                          _Float16* __restrict__ dst) {
  __shared__ float s[16][NP];
  const int lane = threadIdx.x, hf = lane >> 4, lm = lane & 15;
  const int wbase = blockIdx.x * 16;
  const size_t sl = (size_t)blockIdx.y * NP * 32;
  v16h a;
#pragma unroll
  for (int i = 0; i < 16; ++i) a[i] = xth[sl + (size_t)(wbase + lm) * 32 + a_k(hf, i)];
  const float scl = 0.17677669529663687f; // 1/sqrt(32)
  for (int vt = 0; vt < NP / 16; ++vt) {
    v16h bm;
#pragma unroll
    for (int i = 0; i < 16; ++i) bm[i] = xth[sl + (size_t)(vt * 16 + lm) * 32 + 16 * hf + i];
    v8f c = {};
    c = wmma_f16(a, bm, c);
#pragma unroll
    for (int r = 0; r < 8; ++r) s[r + 8 * hf][vt * 16 + lm] = c[r] * scl;
  }
  __syncthreads();
  if (lane < 16) {
    float mx = -1e30f;
    for (int v = 0; v < 500; ++v) mx = fmaxf(mx, s[lane][v]);
    float sum = 0;
    for (int v = 0; v < 500; ++v) { float e = __expf(s[lane][v] - mx); s[lane][v] = e; sum += e; }
    float inv = 1.0f / sum;
    for (int v = 0; v < 500; ++v) s[lane][v] *= inv;
    for (int v = 500; v < NP; ++v) s[lane][v] = 0.0f;
  }
  __syncthreads();
  v8f a0 = {}, a1 = {};
  for (int ks = 0; ks < NP; ks += 32) {
    v16h pa, b0, b1;
#pragma unroll
    for (int i = 0; i < 16; ++i) pa[i] = (_Float16)s[lm][ks + a_k(hf, i)];
#pragma unroll
    for (int i = 0; i < 16; ++i) {
      int k = ks + 16 * hf + i;
      b0[i] = vsrc[sl + (size_t)k * 32 + lm];
      b1[i] = vsrc[sl + (size_t)k * 32 + 16 + lm];
    }
    a0 = wmma_f16(pa, b0, a0);
    a1 = wmma_f16(pa, b1, a1);
  }
#pragma unroll
  for (int r = 0; r < 8; ++r) {
    int w = wbase + r + 8 * hf;
    _Float16 z0 = (w < 500) ? (_Float16)a0[r] : (_Float16)0.0f;
    _Float16 z1 = (w < 500) ? (_Float16)a1[r] : (_Float16)0.0f;
    dst[sl + (size_t)w * 32 + lm]      = z0;
    dst[sl + (size_t)w * 32 + 16 + lm] = z1;
  }
}

// ======== mlp: K=160 gather (h3 + z1..z4), Cout=64, fused gate -> g (b,n,t,32) ========
__global__ void k_mlp(const float* __restrict__ h3, const _Float16* __restrict__ z1,
                      const _Float16* __restrict__ z2, const _Float16* __restrict__ z3,
                      const _Float16* __restrict__ z4, const float* __restrict__ W,
                      const float* __restrict__ bias, float* __restrict__ g) {
  const int lane = threadIdx.x, hf = lane >> 4, lm = lane & 15;
  const int rowbase = blockIdx.x * 16;
  const int row = rowbase + lm;
  const int b = row / 5000, rem = row % 5000, n = rem / 10, t = rem % 10;
  const _Float16* zs[4] = {z1, z2, z3, z4};
  const size_t zoff = ((size_t)(b * 10 + t) * NP + n) * 32;
  v8f acc[4] = {};
#pragma unroll
  for (int ks = 0; ks < 160; ks += 32) {
    v16h a;
#pragma unroll
    for (int i = 0; i < 16; ++i) {
      int k = ks + a_k(hf, i);
      float val;
      if (k < 32) val = h3[(size_t)row * 32 + k];
      else { int zi = (k - 32) >> 5; int c = k & 31; val = (float)zs[zi][zoff + c]; }
      a[i] = (_Float16)val;
    }
    for (int nt = 0; nt < 4; ++nt) {
      v16h bm;
#pragma unroll
      for (int i = 0; i < 16; ++i)
        bm[i] = (_Float16)W[(nt * 16 + lm) * 160 + ks + 16 * hf + i];
      acc[nt] = wmma_f16(a, bm, acc[nt]);
    }
  }
#pragma unroll
  for (int r = 0; r < 8; ++r) {
    int m = rowbase + r + 8 * hf;
    float t0 = tanhf(acc[0][r] + bias[lm]);
    float s0 = 1.0f / (1.0f + __expf(-(acc[2][r] + bias[32 + lm])));
    g[(size_t)m * 32 + lm] = t0 * s0;
    float t1 = tanhf(acc[1][r] + bias[16 + lm]);
    float s1 = 1.0f / (1.0f + __expf(-(acc[3][r] + bias[48 + lm])));
    g[(size_t)m * 32 + 16 + lm] = t1 * s1;
  }
}

// ======== ct1: 10 -> 6 over time; g2 (b,c,n,6) ========
__global__ void k_ct1(const float* __restrict__ g, const float* __restrict__ W,
                      const float* __restrict__ bias, float* __restrict__ g2) {
  int idx = blockIdx.x * blockDim.x + threadIdx.x;
  if (idx >= BB * 32 * NN) return;
  int n = idx % 500, c = (idx / 500) % 32, b = idx / 16000;
  float gv[10];
#pragma unroll
  for (int t = 0; t < 10; ++t) gv[t] = g[(((size_t)b * 500 + n) * 10 + t) * 32 + c];
#pragma unroll
  for (int s = 0; s < 6; ++s) {
    float a = bias[s];
#pragma unroll
    for (int t = 0; t < 10; ++t) a += gv[t] * W[s * 10 + t];
    g2[((size_t)(b * 32 + c) * 500 + n) * 6 + s] = a;
  }
}

__global__ void k_f1(const float* __restrict__ g2, const float* __restrict__ c1W,
                     float* __restrict__ f1) {
  int idx = blockIdx.x * blockDim.x + threadIdx.x;
  if (idx >= BB * 6 * NN) return;
  int n = idx % 500, t = (idx / 500) % 6, b = idx / 3000;
  float a = 0;
  for (int c = 0; c < 32; ++c) a += g2[((size_t)(b * 32 + c) * 500 + n) * 6 + t] * c1W[c];
  f1[(size_t)(b * 6 + t) * 500 + n] = a;
}

__global__ void k_f2(const float* __restrict__ g2, const float* __restrict__ c2W,
                     float* __restrict__ f2) {
  int idx = blockIdx.x * blockDim.x + threadIdx.x;
  if (idx >= BB * 32 * 6) return;
  int t = idx % 6, c = (idx / 6) % 32, b = idx / 192;
  float a = 0;
  for (int n = 0; n < 500; ++n) a += g2[((size_t)(b * 32 + c) * 500 + n) * 6 + t] * c2W[n];
  f2[idx] = a;
}

__global__ void k_m1(const float* __restrict__ f1, const float* __restrict__ tw,
                     float* __restrict__ m1) {
  int idx = blockIdx.x * blockDim.x + threadIdx.x;
  if (idx >= BB * 6 * 32) return;
  int c = idx % 32, t = (idx / 32) % 6, b = idx / 192;
  float a = 0;
  for (int n = 0; n < 500; ++n) a += f1[(size_t)(b * 6 + t) * 500 + n] * tw[n * 32 + c];
  m1[(size_t)(b * 6 + t) * 32 + c] = a;
}

// ======== TATT tail in one block of 576 threads ========
__global__ void k_tatt(const float* __restrict__ m1, const float* __restrict__ f2,
                       const float* __restrict__ tb, const float* __restrict__ tv,
                       const float* __restrict__ g1, const float* __restrict__ b1,
                       float* __restrict__ Tc) {
  __shared__ float sig[576], lg[576], nl[576], cf[576], mu[6], rstd[6];
  int tid = threadIdx.x;
  {
    int b = tid / 36, t = (tid / 6) % 6, ss = tid % 6;
    float a = 0;
    for (int c = 0; c < 32; ++c) a += m1[(b * 6 + t) * 32 + c] * f2[(b * 32 + c) * 6 + ss];
    a += tb[t * 6 + ss];
    sig[tid] = 1.0f / (1.0f + __expf(-a));
  }
  __syncthreads();
  {
    int b = tid / 36, p = (tid / 6) % 6, ss = tid % 6;
    float a = 0;
    for (int t = 0; t < 6; ++t) a += tv[p * 6 + t] * sig[(b * 6 + t) * 6 + ss];
    lg[tid] = a;
  }
  __syncthreads();
  if (tid < 6) {
    float s1 = 0, s2 = 0;
    for (int bp = 0; bp < 96; ++bp) { float v = lg[bp * 6 + tid]; s1 += v; s2 += v * v; }
    float m = s1 / 96.0f;
    mu[tid] = m;
    rstd[tid] = rsqrtf(s2 / 96.0f - m * m + 1e-5f);
  }
  __syncthreads();
  {
    int ss = tid % 6;
    nl[tid] = (lg[tid] - mu[ss]) * rstd[ss] * g1[ss] + b1[ss];
  }
  __syncthreads();
  if (tid < 96) {
    float mx = -1e30f;
    for (int ss = 0; ss < 6; ++ss) mx = fmaxf(mx, nl[tid * 6 + ss]);
    float sum = 0;
    for (int ss = 0; ss < 6; ++ss) { float e = __expf(nl[tid * 6 + ss] - mx); cf[tid * 6 + ss] = e; sum += e; }
    float inv = 1.0f / sum;
    for (int ss = 0; ss < 6; ++ss) cf[tid * 6 + ss] *= inv;
  }
  __syncthreads();
  {
    int b = tid / 36, l = (tid / 6) % 6, qq = tid % 6;
    Tc[tid] = cf[(b * 6 + qq) * 6 + l];  // Tc[b,l,q] = coefs[b,q,l]
  }
}

__global__ void k_zero64(float* __restrict__ a) { if (threadIdx.x < 64) a[threadIdx.x] = 0.0f; }

// ======== xo = leaky(g2 @ Tc) + residual, accumulate per-channel BN2 stats ========
__global__ void k_xo(const float* __restrict__ g2, const float* __restrict__ Tc,
                     const float* __restrict__ res, float* __restrict__ xo,
                     float* __restrict__ accum) {
  __shared__ float ls[256], lq[256];
  int c = blockIdx.x;
  int j = blockIdx.y * 256 + threadIdx.x;
  float val = 0.0f;
  if (j < 48000) {
    int qq = j % 6, n = (j / 6) % 500, b = j / 3000;
    float a = 0;
#pragma unroll
    for (int l = 0; l < 6; ++l)
      a += g2[((size_t)(b * 32 + c) * 500 + n) * 6 + l] * Tc[(b * 6 + l) * 6 + qq];
    a = a > 0.0f ? a : 0.01f * a;
    a += res[(((size_t)b * 500 + n) * 10 + 4 + qq) * 32 + c];
    xo[((size_t)(b * 32 + c) * 500 + n) * 6 + qq] = a;
    val = a;
  }
  ls[threadIdx.x] = val; lq[threadIdx.x] = val * val;
  __syncthreads();
  for (int off = 128; off > 0; off >>= 1) {
    if (threadIdx.x < off) { ls[threadIdx.x] += ls[threadIdx.x + off]; lq[threadIdx.x] += lq[threadIdx.x + off]; }
    __syncthreads();
  }
  if (threadIdx.x == 0) { atomicAdd(&accum[c], ls[0]); atomicAdd(&accum[32 + c], lq[0]); }
}

__global__ void k_final(const float* __restrict__ xo, const float* __restrict__ accum,
                        const float* __restrict__ g, const float* __restrict__ b,
                        float* __restrict__ out) {
  int idx = blockIdx.x * blockDim.x + threadIdx.x;
  if (idx >= BB * 32 * NN * 6) return;
  int c = (idx / 3000) % 32;
  float mu = accum[c] * (1.0f / 48000.0f);
  float var = accum[32 + c] * (1.0f / 48000.0f) - mu * mu;
  out[idx] = (xo[idx] - mu) * rsqrtf(var + 1e-5f) * g[c] + b[c];
}

// ---------------------------------------------------------------------------
extern "C" void kernel_launch(void* const* d_in, const int* in_sizes, int n_in,
                              void* d_out, int out_size, void* d_ws, size_t ws_size,
                              hipStream_t stream) {
  const float* x       = (const float*)d_in[0];
  const float* A       = (const float*)d_in[1];
  const float* conv1_W = (const float*)d_in[2];
  const float* conv1_b = (const float*)d_in[3];
  const float* te1_W   = (const float*)d_in[4];
  const float* te1_b   = (const float*)d_in[5];
  const float* convq_W = (const float*)d_in[6];
  const float* convq_b = (const float*)d_in[7];
  const float* convk_W = (const float*)d_in[8];
  const float* convk_b = (const float*)d_in[9];
  const float* linv_W  = (const float*)d_in[10];
  const float* linv_b  = (const float*)d_in[11];
  const float* lino_W  = (const float*)d_in[12];
  const float* lino_b  = (const float*)d_in[13];
  const float* te2_W   = (const float*)d_in[14];
  const float* te2_b   = (const float*)d_in[15];
  const float* mlp_W   = (const float*)d_in[16];
  const float* mlp_b   = (const float*)d_in[17];
  const float* ct1_W   = (const float*)d_in[18];
  const float* ct1_b   = (const float*)d_in[19];
  const float* tat_c1  = (const float*)d_in[20];
  const float* tat_c2  = (const float*)d_in[21];
  const float* tat_w   = (const float*)d_in[22];
  const float* tat_b   = (const float*)d_in[23];
  const float* tat_v   = (const float*)d_in[24];
  const float* bn1_g   = (const float*)d_in[25];
  const float* bn1_b   = (const float*)d_in[26];
  const float* bn2_g   = (const float*)d_in[27];
  const float* bn2_b   = (const float*)d_in[28];

  char* ws = (char*)d_ws;
  float* res   = (float*)(ws + OFF_RES);
  float* h1    = (float*)(ws + OFF_W1);   // -> o_att -> g2
  float* q     = (float*)(ws + OFF_W2);   // -> h2
  float* k     = (float*)(ws + OFF_W3);   // -> h3
  float* v     = (float*)(ws + OFF_W4);   // -> g -> xo_pre
  _Float16* xth = (_Float16*)(ws + OFF_XTH);
  _Float16* Ah  = (_Float16*)(ws + OFF_AH);
  _Float16* z1  = (_Float16*)(ws + OFF_Z1);
  _Float16* z2  = (_Float16*)(ws + OFF_Z2);
  _Float16* z3  = (_Float16*)(ws + OFF_Z3);
  _Float16* z4  = (_Float16*)(ws + OFF_Z4);
  float* f1    = (float*)(ws + OFF_F1);
  float* f2    = (float*)(ws + OFF_F2);
  float* m1    = (float*)(ws + OFF_M1);
  float* Tc    = (float*)(ws + OFF_TC);
  float* accum = (float*)(ws + OFF_ACC);

  float* o_att = h1;   // reuse after h1 consumed
  float* h2 = q;       // reuse after q consumed
  float* h3 = k;       // reuse after k consumed
  float* g  = v;       // reuse after v consumed
  float* g2 = h1;      // reuse after o_att consumed
  float* xo = v;       // reuse after g consumed

  const int MT = NROW / 16;  // 5000 M-tiles

  k_c1x1_x<<<MT, 32, 0, stream>>>(x, conv1_W, conv1_b, res, 32);
  k_c1x1_x<<<MT, 32, 0, stream>>>(x, te1_W, te1_b, h1, 64);
  k_convt3<<<MT, 32, 0, stream>>>(h1, convq_W, convq_b, q);
  k_convt3<<<MT, 32, 0, stream>>>(h1, convk_W, convk_b, k);
  k_gemm_rm<<<MT, 32, 0, stream>>>(h1, linv_W, linv_b, v, 64, 64);
  k_mhsa<<<(BB * NN * 80 + 255) / 256, 256, 0, stream>>>(q, k, v, o_att);
  k_gemm_rm<<<MT, 32, 0, stream>>>(o_att, lino_W, lino_b, h2, 64, 64);
  k_gemm_rm<<<MT, 32, 0, stream>>>(h2, te2_W, te2_b, h3, 64, 32);
  k_make_xth<<<(NSLICE * NP * 32 + 255) / 256, 256, 0, stream>>>(h3, xth);
  k_make_ah<<<(512 * 512 + 255) / 256, 256, 0, stream>>>(A, Ah);
  k_diffuse<<<NSLICE, 256, 0, stream>>>(Ah, xth, z1);
  k_diffuse<<<NSLICE, 256, 0, stream>>>(Ah, z1, z2);
  k_gcn_att<<<dim3(NP / 16, NSLICE), 32, 0, stream>>>(xth, xth, z3);
  k_gcn_att<<<dim3(NP / 16, NSLICE), 32, 0, stream>>>(xth, z3, z4);
  k_mlp<<<MT, 32, 0, stream>>>(h3, z1, z2, z3, z4, mlp_W, mlp_b, g);
  k_ct1<<<(BB * 32 * NN + 255) / 256, 256, 0, stream>>>(g, ct1_W, ct1_b, g2);
  k_f1<<<(BB * 6 * NN + 255) / 256, 256, 0, stream>>>(g2, tat_c1, f1);
  k_f2<<<(BB * 32 * 6 + 255) / 256, 256, 0, stream>>>(g2, tat_c2, f2);
  k_m1<<<(BB * 6 * 32 + 255) / 256, 256, 0, stream>>>(f1, tat_w, m1);
  k_tatt<<<1, 576, 0, stream>>>(m1, f2, tat_b, tat_v, bn1_g, bn1_b, Tc);
  k_zero64<<<1, 64, 0, stream>>>(accum);
  k_xo<<<dim3(32, (48000 + 255) / 256), 256, 0, stream>>>(g2, Tc, res, xo, accum);
  k_final<<<(BB * 32 * NN * 6 + 255) / 256, 256, 0, stream>>>(xo, accum, bn2_g, bn2_b, (float*)d_out);

  (void)in_sizes; (void)n_in; (void)out_size; (void)ws_size;
}
